// BinaryTTN_17635135717919
// MI455X (gfx1250) — compile-verified, matrix operand order checked by reading, not run on hardware
//
#include <hip/hip_runtime.h>

// ---------------------------------------------------------------------------
// BinaryTTN forward on MI455X (gfx1250, wave32).
// One fused kernel: each workgroup owns a 32-batch tile; the entire 10-layer
// binary tree lives in 256KB of LDS (bf16 cells). Node contractions run on
// v_wmma_f32_16x16x32_bf16: out[m,b] = sum_ij P[m,ij] w[b,i,j], K=256 as
// 8 WMMAs. B fragments are perm-packed once per node into 64 VGPRs and
// reused across the two 16-row batch subtiles (16 WMMAs per node).
// A fragments are built with v_pk_mul_bf16 + OPSEL broadcast: ONE VALU op
// per packed A dword (8 VALU per WMMA).
// ---------------------------------------------------------------------------

typedef __attribute__((ext_vector_type(16))) __bf16       v16bf;
typedef __attribute__((ext_vector_type(8)))  float        v8f;
typedef __attribute__((ext_vector_type(8)))  unsigned int v8u;

#define TTN_WAVES   8
#define TTN_TILE    32           // batch per workgroup (2 x 16-row subtiles)
#define SLOT_BYTES  1024         // 32 batch x 16 bond x bf16 per tree cell
#define NSLOTS      256          // base grid = layer-1 output grid (16x16)
#define TTN_EPS     1e-9f

// ---- bf16 helpers ---------------------------------------------------------
// Truncating pack of two f32 -> bf16 pair in ONE v_perm_b32:
// result = { hi[31:16], lo[31:16] }.
__device__ __forceinline__ unsigned int pk_bf16_tr(float lo, float hi) {
  return __builtin_amdgcn_perm(__builtin_bit_cast(unsigned int, hi),
                               __builtin_bit_cast(unsigned int, lo),
                               0x07060302u);
}
__device__ __forceinline__ float bf_lo(unsigned int u) {
  return __builtin_bit_cast(float, u << 16);
}
__device__ __forceinline__ float bf_hi(unsigned int u) {
  return __builtin_bit_cast(float, u & 0xFFFF0000u);
}

// Packed bf16 multiplies with OPSEL broadcast of one half of src0:
// _ll: (a.lo*b.lo, a.lo*b.hi)   _hh: (a.hi*b.lo, a.hi*b.hi)
__device__ __forceinline__ unsigned int pk_mul_bf16_ll(unsigned int a,
                                                       unsigned int b) {
  unsigned int d;
  asm("v_pk_mul_bf16 %0, %1, %2 op_sel:[0,0] op_sel_hi:[0,1]"
      : "=v"(d) : "v"(a), "v"(b));
  return d;
}
__device__ __forceinline__ unsigned int pk_mul_bf16_hh(unsigned int a,
                                                       unsigned int b) {
  unsigned int d;
  asm("v_pk_mul_bf16 %0, %1, %2 op_sel:[1,0] op_sel_hi:[1,1]"
      : "=v"(d) : "v"(a), "v"(b));
  return d;
}

// ---------------------------------------------------------------------------
// B fragments for one node, all 8 K-blocks, packed bf16 (64 VGPRs).
// A (16x32 bf16) ISA layout: lane half selects K subsets; dword t = K pair.
// WMMA block k: kappa -> i = 2k + (kappa>=16), j = j0 + (kappa%8 pattern).
// B mirrors A with lane = output column n: dwords 0-3 = w[n,2k,j0..j0+7],
// dwords 4-7 = w[n,2k+1,j0..j0+7], consecutive j packed pairwise.
// ---------------------------------------------------------------------------
__device__ __forceinline__ void load_packB(const float* __restrict__ wnode,
                                           int n, int j0, v8u B[8]) {
#pragma unroll
  for (int k = 0; k < 8; ++k) {
    const float* wr = wnode + (n * 16 + 2 * k) * 16 + j0;
    float4 a0 = *(const float4*)(wr);        // w[n,2k,   j0..j0+3]
    float4 a1 = *(const float4*)(wr + 4);    // w[n,2k,   j0+4..j0+7]
    float4 b0 = *(const float4*)(wr + 16);   // w[n,2k+1, j0..j0+3]
    float4 b1 = *(const float4*)(wr + 20);   // w[n,2k+1, j0+4..j0+7]
    B[k][0] = pk_bf16_tr(a0.x, a0.y);
    B[k][1] = pk_bf16_tr(a0.z, a0.w);
    B[k][2] = pk_bf16_tr(a1.x, a1.y);
    B[k][3] = pk_bf16_tr(a1.z, a1.w);
    B[k][4] = pk_bf16_tr(b0.x, b0.y);
    B[k][5] = pk_bf16_tr(b0.z, b0.w);
    B[k][6] = pk_bf16_tr(b1.x, b1.y);
    B[k][7] = pk_bf16_tr(b1.z, b1.w);
  }
}

// A fragment built on the fly: uL[k] = packed (L[2k], L[2k+1]) bf16,
// rP[t] = packed (R[j0+2t], R[j0+2t+1]) bf16.  8 pk_mul per WMMA, K=256.
__device__ __forceinline__ v8f wmma_accum(const unsigned int uL[8],
                                          const unsigned int rP[4],
                                          const v8u B[8], v8f c) {
#pragma unroll
  for (int k = 0; k < 8; ++k) {
    v8u au;
#pragma unroll
    for (int t = 0; t < 4; ++t) {
      au[t]     = pk_mul_bf16_ll(uL[k], rP[t]);  // i = 2k   row of P
      au[4 + t] = pk_mul_bf16_hh(uL[k], rP[t]);  // i = 2k+1 row of P
    }
    c = __builtin_amdgcn_wmma_f32_16x16x32_bf16(
        false, __builtin_bit_cast(v16bf, au),
        false, __builtin_bit_cast(v16bf, B[k]), (short)0, c, false, false);
  }
  return c;
}

// ---------------------------------------------------------------------------
// Normalize D over bond (N) dim, accumulate log-norm, store bf16 rows.
// C/D layout: lane L holds column n = L%16, rows m = r + (L<16 ? 0 : 8).
// ---------------------------------------------------------------------------
__device__ __forceinline__ void finish_node(v8f c, char* slot, float* logacc,
                                            int n, int hf, int sub) {
  float s[8];
#pragma unroll
  for (int r = 0; r < 8; ++r) s[r] = c[r] * c[r];
#pragma unroll
  for (int msk = 1; msk < 16; msk <<= 1) {
#pragma unroll
    for (int r = 0; r < 8; ++r) s[r] += __shfl_xor(s[r], msk, 32);
  }
  float nrm[8], inv[8];
#pragma unroll
  for (int r = 0; r < 8; ++r) {
    nrm[r] = fmaxf(__builtin_amdgcn_sqrtf(s[r]), TTN_EPS);
    inv[r] = __builtin_amdgcn_rcpf(nrm[r]);
  }
  if (n == 0) {  // one lane per half logs its 8 rows
#pragma unroll
    for (int r = 0; r < 8; ++r)
      atomicAdd(&logacc[sub * 16 + hf * 8 + r], __logf(nrm[r]));
  }
#pragma unroll
  for (int r = 0; r < 8; ++r) {
    unsigned int hb =
        __builtin_bit_cast(unsigned int, c[r] * inv[r]) >> 16;
    *(unsigned short*)(slot + (sub * 16 + hf * 8 + r) * 32 + n * 2) =
        (unsigned short)hb;
  }
}

// LDS row reads: cell layout = 32 rows (batch) x 32B (16 bonds bf16).
__device__ __forceinline__ void load_rowL(const char* slot, int row,
                                          unsigned int uL[8]) {
  const uint4* p = (const uint4*)(slot + row * 32);
  uint4 a = p[0], b = p[1];
  uL[0] = a.x; uL[1] = a.y; uL[2] = a.z; uL[3] = a.w;
  uL[4] = b.x; uL[5] = b.y; uL[6] = b.z; uL[7] = b.w;
}
// R row stays packed bf16 (consumed directly by v_pk_mul_bf16).
__device__ __forceinline__ void load_rowR_pk(const char* slot, int row, int j0,
                                             unsigned int rP[4]) {
  const uint4* p = (const uint4*)(slot + row * 32 + j0 * 2);
  uint4 a = p[0];
  rP[0] = a.x; rP[1] = a.y; rP[2] = a.z; rP[3] = a.w;
}

// ---------------------------------------------------------------------------
// Layer-0 cell (fused): H1 cell (X,Y) on the 16x32 grid, one batch subtile.
// Lane (m, hf) produces normalized v[8] = H1[batch, hf*8 .. hf*8+7].
// ---------------------------------------------------------------------------
__device__ __forceinline__ void h1_cell(const float* __restrict__ xin,
                                        const float* __restrict__ w0,
                                        int X, int Y, int batch, int hf,
                                        int logidx, float v[8],
                                        float* logacc) {
  const float* xb = xin + batch * 2048 + X * 64 + Y;  // x[n,c,2X,Y]
  float l0 = xb[0], l1 = xb[1024], r0 = xb[32], r1 = xb[1024 + 32];
  const float* wb = w0 + (X * 32 + Y) * 64 + hf * 32;  // w0[X,Y,b=hf*8..,:,:]
  float ss = 0.f;
#pragma unroll
  for (int t = 0; t < 8; ++t) {
    float4 w = *(const float4*)(wb + t * 4);
    float o = w.x * l0 * r0 + w.y * l0 * r1 + w.z * l1 * r0 + w.w * l1 * r1;
    v[t] = o;
    ss += o * o;
  }
  ss += __shfl_xor(ss, 16, 32);
  float nrm = fmaxf(__builtin_amdgcn_sqrtf(ss), TTN_EPS);
  if (hf == 0) atomicAdd(&logacc[logidx], __logf(nrm));
  float inv = __builtin_amdgcn_rcpf(nrm);
#pragma unroll
  for (int t = 0; t < 8; ++t) v[t] *= inv;
}

// ---------------------------------------------------------------------------
__global__ void __launch_bounds__(256, 1)
ttn_wmma_kernel(const float* __restrict__ xin,
                const float* __restrict__ w0, const float* __restrict__ w1,
                const float* __restrict__ w2, const float* __restrict__ w3,
                const float* __restrict__ w4, const float* __restrict__ w5,
                const float* __restrict__ w6, const float* __restrict__ w7,
                const float* __restrict__ w8, const float* __restrict__ w9,
                float* __restrict__ out) {
  extern __shared__ char smem[];
  char*  hbuf   = smem;                          // 256 cells * 1KB = 256KB
  float* logacc = (float*)(smem + NSLOTS * SLOT_BYTES);  // 32 floats

  const int tid  = threadIdx.x;
  const int wave = tid >> 5;
  const int lane = tid & 31;
  const int m    = lane & 15;   // batch row within subtile / bond column
  const int hf   = lane >> 4;
  const int j0   = hf * 8;
  const int blockBase = blockIdx.x * TTN_TILE;

  if (tid < 32) logacc[tid] = 0.f;
  __syncthreads();

  // ---- Layer 1 (layer 0 fused in): 256 nodes on 16x16 base grid ----------
  for (int node = wave; node < 256; node += TTN_WAVES) {
    int x = node >> 4, y = node & 15;
    v8u B[8];
    load_packB(w1 + node * 4096, m, j0, B);      // reused by both subtiles
#pragma unroll
    for (int sub = 0; sub < 2; ++sub) {
      int batch = blockBase + sub * 16 + m;
      float lv[8], rv[8];
      h1_cell(xin, w0, x, 2 * y,     batch, hf, sub * 16 + m, lv, logacc);
      h1_cell(xin, w0, x, 2 * y + 1, batch, hf, sub * 16 + m, rv, logacc);
      float lo[8];
#pragma unroll
      for (int t = 0; t < 8; ++t) lo[t] = __shfl_xor(lv[t], 16, 32);
      unsigned int uL[8], rP[4];
#pragma unroll
      for (int k = 0; k < 4; ++k) {
        // bonds 0..7 live in half 0, bonds 8..15 in half 1
        float e0 = hf ? lo[2 * k]     : lv[2 * k];
        float e1 = hf ? lo[2 * k + 1] : lv[2 * k + 1];
        float f0 = hf ? lv[2 * k]     : lo[2 * k];
        float f1 = hf ? lv[2 * k + 1] : lo[2 * k + 1];
        uL[k]     = pk_bf16_tr(e0, e1);
        uL[4 + k] = pk_bf16_tr(f0, f1);
        rP[k]     = pk_bf16_tr(rv[2 * k], rv[2 * k + 1]);
      }
      v8f c;
#pragma unroll
      for (int i = 0; i < 8; ++i) c[i] = 0.f;
      c = wmma_accum(uL, rP, B, c);
      finish_node(c, hbuf + node * SLOT_BYTES, logacc, m, hf, sub);
    }
  }
  __syncthreads();

  // ---- Layers 2..8: in-place tree reduction over the 16x16 base grid -----
  const float* const wl[7] = {w2, w3, w4, w5, w6, w7, w8};
  const int NA[7] = {128, 64, 32, 16, 8, 4, 2};
  const int GW[7] = {16, 8, 8, 4, 4, 2, 2};
  const int EH[7] = {2, 2, 4, 4, 8, 8, 16};
  const int EW[7] = {1, 2, 2, 4, 4, 8, 8};
  const int RO[7] = {16, 1, 32, 2, 64, 4, 128};
#pragma unroll
  for (int l = 0; l < 7; ++l) {
    const float* w = wl[l];
    for (int node = wave; node < NA[l]; node += TTN_WAVES) {
      int x = node / GW[l], y = node % GW[l];
      int ps = x * EH[l] * 16 + y * EW[l];
      char* sl = hbuf + ps * SLOT_BYTES;             // left child (reused)
      char* sr = hbuf + (ps + RO[l]) * SLOT_BYTES;   // right child
      v8u B[8];
      load_packB(w + node * 4096, m, j0, B);
#pragma unroll
      for (int sub = 0; sub < 2; ++sub) {
        unsigned int uL[8], rP[4];
        load_rowL(sl, sub * 16 + m, uL);
        load_rowR_pk(sr, sub * 16 + m, j0, rP);
        v8f c;
#pragma unroll
        for (int i = 0; i < 8; ++i) c[i] = 0.f;
        c = wmma_accum(uL, rP, B, c);
        finish_node(c, sl, logacc, m, hf, sub);      // in-place overwrite
      }
    }
    __syncthreads();
  }

  // ---- Layer 9: single node, bond_out = 1 (scalar) ------------------------
  if (wave == 0) {
#pragma unroll
    for (int sub = 0; sub < 2; ++sub) {
      unsigned int uL[8], rP[4];
      load_rowL(hbuf, sub * 16 + m, uL);                      // left  = slot 0
      load_rowR_pk(hbuf + 8 * SLOT_BYTES, sub * 16 + m, j0, rP);  // right
      float r[8];
#pragma unroll
      for (int t = 0; t < 4; ++t) {
        r[2 * t] = bf_lo(rP[t]);
        r[2 * t + 1] = bf_hi(rP[t]);
      }
      float acc = 0.f;
#pragma unroll
      for (int k = 0; k < 8; ++k) {
        float s0 = 0.f, s1 = 0.f;
#pragma unroll
        for (int t = 0; t < 8; ++t) {
          s0 += w9[(2 * k) * 16 + j0 + t] * r[t];
          s1 += w9[(2 * k + 1) * 16 + j0 + t] * r[t];
        }
        acc += bf_lo(uL[k]) * s0 + bf_hi(uL[k]) * s1;
      }
      acc += __shfl_xor(acc, 16, 32);  // combine j halves
      float nrm = fmaxf(fabsf(acc), TTN_EPS);
      if (hf == 0) {
        int batch = blockBase + sub * 16 + m;
        out[batch]        = acc * __builtin_amdgcn_rcpf(nrm);  // final h
        out[8192 + batch] = logacc[sub * 16 + m] + __logf(nrm);
      }
    }
  }
}

// ---------------------------------------------------------------------------
extern "C" void kernel_launch(void* const* d_in, const int* in_sizes, int n_in,
                              void* d_out, int out_size, void* d_ws,
                              size_t ws_size, hipStream_t stream) {
  (void)in_sizes; (void)n_in; (void)out_size; (void)d_ws; (void)ws_size;
  const float* xin = (const float*)d_in[0];
  const float* w[10];
  for (int i = 0; i < 10; ++i) w[i] = (const float*)d_in[1 + i];
  float* out = (float*)d_out;

  const size_t shmem = NSLOTS * SLOT_BYTES + 32 * sizeof(float);  // ~256KB
  (void)hipFuncSetAttribute((const void*)ttn_wmma_kernel,
                            hipFuncAttributeMaxDynamicSharedMemorySize,
                            (int)shmem);
  dim3 grid(8192 / TTN_TILE);   // 256 workgroups, one 32-batch tile each
  dim3 block(TTN_WAVES * 32);   // 8 wave32
  ttn_wmma_kernel<<<grid, block, shmem, stream>>>(
      xin, w[0], w[1], w[2], w[3], w[4], w[5], w[6], w[7], w[8], w[9], out);
}
